// RecurrentCrossLinearAttention_4045859193552
// MI455X (gfx1250) — compile-verified, matrix operand order checked by reading, not run on hardware
//
#include <hip/hip_runtime.h>

typedef __bf16 bf16_8  __attribute__((ext_vector_type(8)));
typedef __bf16 bf16_16 __attribute__((ext_vector_type(16)));
typedef float  f32_8   __attribute__((ext_vector_type(8)));

#define S_LEN 4096
#define H_DIM 16
#define D_DIM 64
#define M_DIM 64
#define HD    (H_DIM * D_DIM)   // stride (floats) between consecutive s
#define CH    64                // sequence chunk per LDS stage
#define PITCH 72                // bf16 elements per LDS row: 144B = 9*16 (aligned, conflict-free frag reads)
#define PS    65                // f32 pitch for S_state tile store

// pack two floats -> two bf16 (RNE) in one dword: x -> low half, y -> high half
__device__ __forceinline__ unsigned int pack_bf2(float x, float y) {
    unsigned int ux = __float_as_uint(x);
    unsigned int uy = __float_as_uint(y);
    ux += 0x7FFFu + ((ux >> 16) & 1u);
    uy += 0x7FFFu + ((uy >> 16) & 1u);
    return (ux >> 16) | (uy & 0xFFFF0000u);
}

__device__ __forceinline__ float elu1(float x) {
    // elu(x)+1 = x+1 (x>0) else exp(x)
    return x > 0.0f ? x + 1.0f : __expf(x);
}

__device__ __forceinline__ bf16_16 make_frag(const unsigned short* rowp, int kb) {
    // fragment = two contiguous 8-element bf16 runs: k in [kb, kb+8) and [kb+16, kb+24)
    bf16_8 lo = *(const bf16_8*)(rowp + kb);
    bf16_8 hi = *(const bf16_8*)(rowp + kb + 16);
    bf16_16 r;
#pragma unroll
    for (int i = 0; i < 8; ++i) { r[i] = lo[i]; r[i + 8] = hi[i]; }
    return r;
}

__global__ __launch_bounds__(256)
void RecurrentCrossLinearAttention_kernel(const float* __restrict__ query,
                                          const float* __restrict__ keys,
                                          const float* __restrict__ values,
                                          const float* __restrict__ key_len,
                                          float* __restrict__ out)
{
    __shared__ unsigned short sKt[D_DIM * PITCH];  // K chunk, transposed: [d][s_local]
    __shared__ unsigned short sVt[M_DIM * PITCH];  // V chunk, transposed: [m][s_local]
    __shared__ float sS[M_DIM * PS];               // S_state[m][d]
    __shared__ float sZ[D_DIM];
    __shared__ float sQ[D_DIM];
    __shared__ float sQZ;

    const int tid = threadIdx.x;
    const int nh  = blockIdx.x;              // n*H + h
    const int n   = nh >> 4;
    const int h   = nh & 15;

    const float* kbase = keys    + ((size_t)n * S_LEN * H_DIM + h) * (size_t)D_DIM;
    const float* vbase = values  + ((size_t)n * S_LEN * H_DIM + h) * (size_t)M_DIM;
    const float* lbase = key_len + (size_t)n * S_LEN;

    if (tid < D_DIM) sZ[tid] = 0.0f;

    // ---- loader mapping: thread owns d/m columns 4*(tid&15)..+3 and
    //      s-pairs {2*sp, 2*sp+1} and {+32} within each 64-row chunk ----
    const int dbase = (tid & 15) * 4;
    const int sp    = tid >> 4;

    // ---- compute mapping: 8 waves x 2 tiles; wave w: m0 = 16*(w>>1), d0 = 32*(w&1), +16 ----
    const int wv   = tid >> 5;
    const int lane = tid & 31;
    const int half = lane >> 4;
    const int ln   = lane & 15;
    const int m0   = (wv >> 1) * 16;
    const int d0a  = (wv & 1) * 32;
    const int d0b  = d0a + 16;

    f32_8 accA = {0.f, 0.f, 0.f, 0.f, 0.f, 0.f, 0.f, 0.f};
    f32_8 accB = {0.f, 0.f, 0.f, 0.f, 0.f, 0.f, 0.f, 0.f};
    float z0 = 0.f, z1 = 0.f, z2 = 0.f, z3 = 0.f;

    for (int s0 = 0; s0 < S_LEN; s0 += CH) {
        // -------- global loads into registers (no LDS dependence yet) --------
        float4 kf[4], vf[4];
        float  lf[4];
#pragma unroll
        for (int i = 0; i < 2; ++i) {
            const int sl0 = 2 * sp + 32 * i;
            const int s   = s0 + sl0;
            kf[2*i+0] = *(const float4*)(kbase + (size_t)(s + 0) * HD + dbase);
            kf[2*i+1] = *(const float4*)(kbase + (size_t)(s + 1) * HD + dbase);
            vf[2*i+0] = *(const float4*)(vbase + (size_t)(s + 0) * HD + dbase);
            vf[2*i+1] = *(const float4*)(vbase + (size_t)(s + 1) * HD + dbase);
            lf[2*i+0] = lbase[s + 0];
            lf[2*i+1] = lbase[s + 1];
        }
        if (s0 < S_LEN - 2 * CH) {   // uniform (scalar) guard: next chunk exists
#pragma unroll
            for (int i = 0; i < 2; ++i) {
                const int s = s0 + 2 * sp + 32 * i + CH;
                __builtin_prefetch(kbase + (size_t)s * HD + dbase, 0, 1);
                __builtin_prefetch(vbase + (size_t)s * HD + dbase, 0, 1);
            }
        }

        __syncthreads();   // previous chunk's compute finished reading LDS

        // -------- feature map + packed bf16 convert + transposed LDS store --------
#pragma unroll
        for (int i = 0; i < 2; ++i) {
            const int sl0 = 2 * sp + 32 * i;
            const float l0 = lf[2*i+0], l1 = lf[2*i+1];
            float ka0 = elu1(kf[2*i+0].x) * l0, ka1 = elu1(kf[2*i+1].x) * l1;
            float kb0 = elu1(kf[2*i+0].y) * l0, kb1 = elu1(kf[2*i+1].y) * l1;
            float kc0 = elu1(kf[2*i+0].z) * l0, kc1 = elu1(kf[2*i+1].z) * l1;
            float kd0 = elu1(kf[2*i+0].w) * l0, kd1 = elu1(kf[2*i+1].w) * l1;
            z0 += ka0 + ka1;
            z1 += kb0 + kb1;
            z2 += kc0 + kc1;
            z3 += kd0 + kd1;
            *(unsigned int*)&sKt[(dbase + 0) * PITCH + sl0] = pack_bf2(ka0, ka1);
            *(unsigned int*)&sKt[(dbase + 1) * PITCH + sl0] = pack_bf2(kb0, kb1);
            *(unsigned int*)&sKt[(dbase + 2) * PITCH + sl0] = pack_bf2(kc0, kc1);
            *(unsigned int*)&sKt[(dbase + 3) * PITCH + sl0] = pack_bf2(kd0, kd1);
            *(unsigned int*)&sVt[(dbase + 0) * PITCH + sl0] = pack_bf2(vf[2*i+0].x, vf[2*i+1].x);
            *(unsigned int*)&sVt[(dbase + 1) * PITCH + sl0] = pack_bf2(vf[2*i+0].y, vf[2*i+1].y);
            *(unsigned int*)&sVt[(dbase + 2) * PITCH + sl0] = pack_bf2(vf[2*i+0].z, vf[2*i+1].z);
            *(unsigned int*)&sVt[(dbase + 3) * PITCH + sl0] = pack_bf2(vf[2*i+0].w, vf[2*i+1].w);
        }

        __syncthreads();

        // -------- WMMA: C[m,d] += Sum_k V[s0+k, m] * K[s0+k, d] --------
#pragma unroll
        for (int kk = 0; kk < 2; ++kk) {
            const int kb = kk * 32 + 8 * half;
            bf16_16 afrag = make_frag(&sVt[(m0  + ln) * PITCH], kb);
            bf16_16 b0    = make_frag(&sKt[(d0a + ln) * PITCH], kb);
            bf16_16 b1    = make_frag(&sKt[(d0b + ln) * PITCH], kb);
            accA = __builtin_amdgcn_wmma_f32_16x16x32_bf16(
                       false, afrag, false, b0, (short)0, accA, false, false);
            accB = __builtin_amdgcn_wmma_f32_16x16x32_bf16(
                       false, afrag, false, b1, (short)0, accB, false, false);
        }
    }

    // -------- reduce Z (each thread owns fixed d-range) --------
    atomicAdd(&sZ[dbase + 0], z0);
    atomicAdd(&sZ[dbase + 1], z1);
    atomicAdd(&sZ[dbase + 2], z2);
    atomicAdd(&sZ[dbase + 3], z3);

    // -------- scatter S_state tiles per documented C layout --------
#pragma unroll
    for (int r = 0; r < 8; ++r) {
        const int row = m0 + r + 8 * half;
        sS[row * PS + d0a + ln] = accA[r];
        sS[row * PS + d0b + ln] = accB[r];
    }

    if (tid < D_DIM) sQ[tid] = elu1(query[(size_t)nh * D_DIM + tid]);
    __syncthreads();

    if (tid == 0) {
        float dot = 0.0f;
#pragma unroll
        for (int d = 0; d < D_DIM; ++d) dot += sQ[d] * sZ[d];
        sQZ = 1.0f / (dot + 1e-6f);
    }
    __syncthreads();

    if (tid < M_DIM) {
        float acc = 0.0f;
#pragma unroll
        for (int d = 0; d < D_DIM; ++d) acc += sQ[d] * sS[tid * PS + d];
        out[(size_t)nh * M_DIM + tid] = acc * sQZ;
    }
}

extern "C" void kernel_launch(void* const* d_in, const int* in_sizes, int n_in,
                              void* d_out, int out_size, void* d_ws, size_t ws_size,
                              hipStream_t stream) {
    (void)in_sizes; (void)n_in; (void)d_ws; (void)ws_size; (void)out_size;
    const float* query   = (const float*)d_in[0];
    const float* keys    = (const float*)d_in[1];
    const float* values  = (const float*)d_in[2];
    const float* key_len = (const float*)d_in[3];
    float* out = (float*)d_out;

    const int n_blocks = 16 * H_DIM;   // N * H = 256, one (n,h) per workgroup
    RecurrentCrossLinearAttention_kernel<<<n_blocks, 256, 0, stream>>>(
        query, keys, values, key_len, out);
}
